// MultiHeadedSelfAttention_4509715660804
// MI455X (gfx1250) — matrix operand order repaired and checked by reading, not compiled, not run on hardware
//
#include <hip/hip_runtime.h>
#include <hip/hip_bf16.h>
#include <stdint.h>

typedef __attribute__((ext_vector_type(16))) __bf16 v16bf;
typedef __attribute__((ext_vector_type(8)))  float  v8f;
typedef uint32_t u32x4 __attribute__((ext_vector_type(4)));
typedef int      i32x4 __attribute__((ext_vector_type(4)));
typedef int      i32x8 __attribute__((ext_vector_type(8)));

#define B_  4
#define S_  2048
#define D_  512
#define H_  8
#define HD_ 64

#if __has_builtin(__builtin_amdgcn_tensor_load_to_lds)
#define HAVE_TDM 1
#else
#define HAVE_TDM 0
#endif

__device__ __forceinline__ unsigned short f32_bf16(float f) {
  union { float f; uint32_t u; } a; a.f = f;
  a.u += 0x7fffu + ((a.u >> 16) & 1u);   // round-to-nearest-even
  return (unsigned short)(a.u >> 16);
}

__device__ __forceinline__ uint32_t lds_off(const void* p) {
  // low 32 bits of a flat LDS address are the LDS byte address (ISA 10.2)
  return (uint32_t)(uintptr_t)p;
}

union Frag { v16bf v; uint32_t u[8]; };

// ---------------------------------------------------------------- stage 0
__global__ __launch_bounds__(256) void cvt_bf16(const float* __restrict__ src,
                                                unsigned short* __restrict__ dst, int n) {
  int i = blockIdx.x * blockDim.x + threadIdx.x;
  if (i < n) dst[i] = f32_bf16(src[i]);
}

// ---------------------------------------------------------------- stage 1: fused QKV GEMM
// out[s,e] = sum_d x[s,d]*W[e,d] + b[e]   (M=8192, N=512 per matrix, K=512)
__global__ __launch_bounds__(256) void qkv_gemm(
    const unsigned short* __restrict__ xb, const unsigned short* __restrict__ wb,
    const float* __restrict__ bq, const float* __restrict__ bk, const float* __restrict__ bv,
    unsigned short* __restrict__ qb, unsigned short* __restrict__ kb, unsigned short* __restrict__ vb)
{
  __shared__ uint32_t lx[64 * 256];          // 64 rows x 512 bf16 = 64KB, XOR-swizzled
  const int tid = threadIdx.x;
  const int m0  = blockIdx.x * 64;

  { // stage x tile into LDS, 128-bit copies, uint4-granularity XOR swizzle vs bank conflicts
    const uint4* src = (const uint4*)(xb + (size_t)m0 * D_);
    uint4* dst = (uint4*)lx;
    #pragma unroll
    for (int i = 0; i < 16; ++i) {
      int idx = tid + i * 256;               // 4096 uint4 = 64 rows x 64 cols
      int row = idx >> 6, col = idx & 63;
      dst[(row << 6) | (col ^ (row & 15))] = src[idx];
    }
  }
  __syncthreads();

  const int wave = tid >> 5, lane = tid & 31;
  const int half = lane >> 4, ln = lane & 15;
  const int stripe = wave & 3, chalf = wave >> 2;
  const int arow = stripe * 16 + ln;
  const uint32_t* xrow = lx + arow * 256;    // A-frag row base (swizzled cols)

  for (int t = 0; t < 48; ++t) {                     // 48 16x16 output subtiles per wave
    const int cg  = chalf * 768 + t * 16;            // fused column over [q|k|v]
    const int mat = cg >> 9;
    const int nb  = cg & 511;
    const float* bias = (mat == 0) ? bq : (mat == 1) ? bk : bv;
    const float bvv = bias[nb + ln];
    v8f acc = { bvv, bvv, bvv, bvv, bvv, bvv, bvv, bvv };   // C col = lane&15

    const uint32_t* wg = (const uint32_t*)(wb) + (size_t)mat * (D_ * D_ / 2)
                         + (size_t)(nb + ln) * 256 + 8 * half;
    #pragma unroll 4
    for (int kk = 0; kk < 16; ++kk) {                // K = 512 in chunks of 32
      Frag a, b;
      const int cA = (((kk << 2) + 2 * half) ^ ln) << 2;       // swizzled uint4 col, low 4 dw
      const int cB = (((kk << 2) + 2 * half + 1) ^ ln) << 2;   // swizzled uint4 col, high 4 dw
      #pragma unroll
      for (int v = 0; v < 4; ++v) a.u[v] = xrow[cA + v];
      #pragma unroll
      for (int v = 4; v < 8; ++v) a.u[v] = xrow[cB + (v - 4)];
      #pragma unroll
      for (int v = 0; v < 8; ++v) b.u[v] = wg[kk * 16 + v];
      acc = __builtin_amdgcn_wmma_f32_16x16x32_bf16(false, a.v, false, b.v,
                                                    (short)0, acc, false, false);
    }
    unsigned short* dst = (mat == 0) ? qb : (mat == 1) ? kb : vb;
    const int row0 = m0 + stripe * 16 + 8 * half;
    #pragma unroll
    for (int v = 0; v < 8; ++v)
      dst[(size_t)(row0 + v) * D_ + nb + ln] = f32_bf16(acc[v]);
  }
}

// ---------------------------------------------------------------- stage 2: flash attention
#define KP 36   // K tile row pitch in dwords (64 bf16 + 4 dw pad) -> conflict-free, 16B aligned
#define VP 66   // Vt/P row pitch in halfwords (33 dw, gcd(33,64)=1 -> conflict-free)

__global__ __launch_bounds__(256) void attention(
    const unsigned short* __restrict__ qb, const unsigned short* __restrict__ kb,
    const unsigned short* __restrict__ vb, const int* __restrict__ mask,
    float* __restrict__ out)
{
  __shared__ uint32_t       lK[64 * KP];     // key tile [j][d], padded pitch
  __shared__ unsigned short lVt[64 * VP];    // value tile transposed [e][j], padded
  __shared__ uint32_t       lP[8][16 * (VP / 2)]; // per-wave P tile [m][j], padded

  const int tid  = threadIdx.x;
  const int wave = tid >> 5, lane = tid & 31;
  const int half = lane >> 4, ln = lane & 15;

  const int qt = blockIdx.x & 15;            // 16 query tiles of 128
  const int bh = blockIdx.x >> 4;
  const int b  = bh >> 3, h = bh & 7;

  // Q fragments resident in registers: 16 rows x 64 d per wave (2 K-chunks)
  Frag aq[2];
  {
    const uint32_t* qg = (const uint32_t*)qb
        + (size_t)(b * S_ + qt * 128 + wave * 16 + ln) * 256 + h * 32 + 4 * half;
    #pragma unroll
    for (int c = 0; c < 2; ++c)
      #pragma unroll
      for (int v = 0; v < 8; ++v)
        aq[c].u[v] = qg[c * 16 + (v < 4 ? v : v + 4)];
  }

  float m_run[8], l_run[8];
  v8f oacc[4];
  #pragma unroll
  for (int v = 0; v < 8; ++v) { m_run[v] = -1e30f; l_run[v] = 0.f; }
  #pragma unroll
  for (int s = 0; s < 4; ++s) oacc[s] = (v8f){0,0,0,0,0,0,0,0};

  for (int jt = 0; jt < 32; ++jt) {
    __syncthreads();

    // ---- K tile via Tensor Data Mover: 64x64 bf16 2D tile, stride 512, LDS pad
    //      pad_interval=32dw (code 4), pad_amount=4dw (code 3) => row pitch KP=36 dw
#if HAVE_TDM
    if (wave == 0) {
      const uint64_t ga = (uint64_t)(uintptr_t)(kb + (size_t)(b * S_ + jt * 64) * D_ + h * 64);
      u32x4 g0 = { 1u,                                  // count=1, user mode
                   lds_off(lK),                         // lds_addr
                   (uint32_t)ga,                        // global_addr[31:0]
                   (uint32_t)(ga >> 32) | (2u << 30) }; // global_addr[56:32] | type=2
      i32x8 g1 = { (int)0x07110000,     // data_size=2B, pad_enable, interval=32dw, amount=4dw
                   (int)(64u << 16),    // tensor_dim0 = 64
                   (int)(64u << 16),    // tensor_dim1 = 64
                   (int)(64u << 16),    // tile_dim0 = 64
                   64,                  // tile_dim1 = 64
                   512,                 // tensor_dim0_stride = 512 elements
                   0, 0 };
      i32x4 z4 = { 0, 0, 0, 0 };
#if __clang_major__ >= 23
      i32x8 z8 = { 0, 0, 0, 0, 0, 0, 0, 0 };
      __builtin_amdgcn_tensor_load_to_lds(g0, g1, z4, z4, z8, 0);
#else
      __builtin_amdgcn_tensor_load_to_lds(g0, g1, z4, z4, 0);
#endif
    }
#else
    { // fallback: manual padded copy, identical layout
      const uint4* kg = (const uint4*)kb;
      uint4* d4 = (uint4*)lK;
      #pragma unroll
      for (int i = 0; i < 2; ++i) {
        int idx = tid + i * 256;                       // 512 x uint4
        int j = idx >> 3, c4 = idx & 7;
        d4[j * (KP / 4) + c4] = kg[(size_t)(b * S_ + jt * 64 + j) * 64 + h * 8 + c4];
      }
    }
#endif

    // ---- V tile transposed (manual: TDM cannot transpose), conflict-free padded pitch
    #pragma unroll
    for (int i = 0; i < 16; ++i) {
      int idx = tid + i * 256;                         // 4096 x bf16, coalesced reads
      int j = idx >> 6, e = idx & 63;
      lVt[e * VP + j] = vb[(size_t)(b * S_ + jt * 64 + j) * D_ + h * 64 + e];
    }

    // ---- prefetch next K/V tiles into cache while we compute this one
    if (jt < 31) {
      const size_t nrow = (size_t)(b * S_ + (jt + 1) * 64);
      const char* nk = (const char*)(kb + nrow * D_ + h * 64);
      const char* nv = (const char*)(vb + nrow * D_ + h * 64);
      const int pr = tid >> 2, pc = (tid & 3) * 32;    // 64 rows x 128B footprint
      __builtin_prefetch(nk + (size_t)pr * 1024 + pc, 0, 0);
      __builtin_prefetch(nv + (size_t)pr * 1024 + pc, 0, 0);
    }

    __builtin_amdgcn_s_wait_tensorcnt(0);
    __syncthreads();

    // ---- scores S = Q K^T : 4 col-subtiles x 2 K-chunks
    v8f sc[4];
    #pragma unroll
    for (int sub = 0; sub < 4; ++sub) {
      v8f c0 = (v8f){0,0,0,0,0,0,0,0};
      #pragma unroll
      for (int c = 0; c < 2; ++c) {
        Frag bk_;
        #pragma unroll
        for (int v = 0; v < 8; ++v)
          bk_.u[v] = lK[(sub * 16 + ln) * KP + c * 16 + 4 * half + (v < 4 ? v : v + 4)];
        c0 = __builtin_amdgcn_wmma_f32_16x16x32_bf16(false, aq[c].v, false, bk_.v,
                                                     (short)0, c0, false, false);
      }
      sc[sub] = c0;
    }

    // ---- scale 1/sqrt(64) + additive mask (per key column)
    #pragma unroll
    for (int sub = 0; sub < 4; ++sub) {
      const int jg = jt * 64 + sub * 16 + ln;
      const float mk = -10000.f * (1.f - (float)mask[b * S_ + jg]);
      #pragma unroll
      for (int v = 0; v < 8; ++v) sc[sub][v] = sc[sub][v] * 0.125f + mk;
    }

    // ---- online softmax: row reductions across 16-lane halves (C-layout rows)
    float mx[8];
    #pragma unroll
    for (int v = 0; v < 8; ++v)
      mx[v] = fmaxf(fmaxf(sc[0][v], sc[1][v]), fmaxf(sc[2][v], sc[3][v]));
    #pragma unroll
    for (int off = 1; off < 16; off <<= 1)
      #pragma unroll
      for (int v = 0; v < 8; ++v)
        mx[v] = fmaxf(mx[v], __shfl_xor(mx[v], off, 32));

    float scl[8], rs[8];
    #pragma unroll
    for (int v = 0; v < 8; ++v) {
      float mn = fmaxf(m_run[v], mx[v]);
      scl[v] = __expf(m_run[v] - mn);
      m_run[v] = mn;
      rs[v] = 0.f;
    }
    #pragma unroll
    for (int sub = 0; sub < 4; ++sub)
      #pragma unroll
      for (int v = 0; v < 8; ++v) {
        float p = __expf(sc[sub][v] - m_run[v]);
        sc[sub][v] = p;
        rs[v] += p;
      }
    #pragma unroll
    for (int off = 1; off < 16; off <<= 1)
      #pragma unroll
      for (int v = 0; v < 8; ++v)
        rs[v] += __shfl_xor(rs[v], off, 32);
    #pragma unroll
    for (int v = 0; v < 8; ++v) l_run[v] = l_run[v] * scl[v] + rs[v];
    #pragma unroll
    for (int sub = 0; sub < 4; ++sub)
      #pragma unroll
      for (int v = 0; v < 8; ++v) oacc[sub][v] *= scl[v];

    // ---- C-layout -> A-layout restripe of P through per-wave LDS (padded pitch)
    unsigned short* pp = (unsigned short*)lP[wave];
    #pragma unroll
    for (int sub = 0; sub < 4; ++sub)
      #pragma unroll
      for (int v = 0; v < 8; ++v)
        pp[(8 * half + v) * VP + sub * 16 + ln] = f32_bf16(sc[sub][v]);
    asm volatile("s_wait_dscnt 0" ::: "memory");   // wave-local LDS RAW fence

    // ---- O += P V
    Frag ap[2];
    #pragma unroll
    for (int c = 0; c < 2; ++c)
      #pragma unroll
      for (int v = 0; v < 8; ++v)
        ap[c].u[v] = lP[wave][ln * (VP / 2) + c * 16 + 4 * half + (v < 4 ? v : v + 4)];
    const uint32_t* lVtU = (const uint32_t*)lVt;
    #pragma unroll
    for (int sub = 0; sub < 4; ++sub) {
      #pragma unroll
      for (int c = 0; c < 2; ++c) {
        Frag bv_;
        #pragma unroll
        for (int v = 0; v < 8; ++v)
          bv_.u[v] = lVtU[(sub * 16 + ln) * (VP / 2) + c * 16 + 4 * half + (v < 4 ? v : v + 4)];
        oacc[sub] = __builtin_amdgcn_wmma_f32_16x16x32_bf16(false, ap[c].v, false, bv_.v,
                                                            (short)0, oacc[sub], false, false);
      }
    }
  }

  // ---- epilogue: normalize and store fp32 output (B,S,D), head h at cols h*64..
  const int srow0 = qt * 128 + wave * 16 + 8 * half;
  #pragma unroll
  for (int sub = 0; sub < 4; ++sub)
    #pragma unroll
    for (int v = 0; v < 8; ++v)
      out[(size_t)(b * S_ + srow0 + v) * D_ + h * 64 + sub * 16 + ln] =
          oacc[sub][v] / l_run[v];
}

// ---------------------------------------------------------------- launcher
extern "C" void kernel_launch(void* const* d_in, const int* in_sizes, int n_in,
                              void* d_out, int out_size, void* d_ws, size_t ws_size,
                              hipStream_t stream) {
  (void)in_sizes; (void)n_in; (void)out_size; (void)ws_size;
  const float* x  = (const float*)d_in[0];
  const int*  msk = (const int*)d_in[1];
  const float* Wq = (const float*)d_in[2];
  const float* bq = (const float*)d_in[3];
  const float* Wk = (const float*)d_in[4];
  const float* bk = (const float*)d_in[5];
  const float* Wv = (const float*)d_in[6];
  const float* bv = (const float*)d_in[7];

  char* ws = (char*)d_ws;
  unsigned short* xb = (unsigned short*)(ws);              // 8 MB  : x bf16
  unsigned short* wb = (unsigned short*)(ws + 8388608);    // 1.5MB : Wq|Wk|Wv bf16
  unsigned short* qb = (unsigned short*)(ws + 9961472);    // 8 MB
  unsigned short* kb = (unsigned short*)(ws + 18350080);   // 8 MB
  unsigned short* vb = (unsigned short*)(ws + 26738688);   // 8 MB

  const int NX = B_ * S_ * D_;        // 4,194,304
  const int NW = D_ * D_;             // 262,144

  cvt_bf16<<<(NX + 255) / 256, 256, 0, stream>>>(x,  xb, NX);
  cvt_bf16<<<(NW + 255) / 256, 256, 0, stream>>>(Wq, wb,           NW);
  cvt_bf16<<<(NW + 255) / 256, 256, 0, stream>>>(Wk, wb + NW,      NW);
  cvt_bf16<<<(NW + 255) / 256, 256, 0, stream>>>(Wv, wb + 2 * NW,  NW);

  qkv_gemm<<<(B_ * S_) / 64, 256, 0, stream>>>(xb, wb, bq, bk, bv, qb, kb, vb);

  attention<<<B_ * H_ * (S_ / 128), 256, 0, stream>>>(qb, kb, vb, msk, (float*)d_out);
}
